// Stage_87428354277718
// MI455X (gfx1250) — matrix operand order repaired and checked by reading, not compiled, not run on hardware
//
#include <hip/hip_runtime.h>
#include <hip/hip_bf16.h>

typedef __bf16 bf16_t;
typedef __attribute__((ext_vector_type(16))) __bf16 v16bf;
typedef __attribute__((ext_vector_type(8)))  float  v8f;
typedef int v4i_g __attribute__((vector_size(16)));

// ---------------- CDNA5 async global->LDS path (guarded) ----------------
#if __has_builtin(__builtin_amdgcn_global_load_async_to_lds_b128)
#define HAVE_ASYNC 1
#define G2L_B128(gptr, lptr)                                                 \
    __builtin_amdgcn_global_load_async_to_lds_b128(                          \
        (v4i_g*)(gptr), (v4i_g*)(lptr), 0, 0)
#else
#define HAVE_ASYNC 0
#endif

__device__ __forceinline__ void async_wait0() {
#if HAVE_ASYNC
#if __has_builtin(__builtin_amdgcn_s_wait_asynccnt)
    __builtin_amdgcn_s_wait_asynccnt(0);
#else
    asm volatile("s_wait_asynccnt 0" ::: "memory");
#endif
#endif
}

// ---------------- problem constants ----------------
#define BB   64
#define HH   56
#define WW_  56
#define DD   128
#define WS7  7
#define SH3  3
#define NHD  4
#define HD   32
#define NG2  8
#define NG   64
#define NTOK 49

static constexpr long M_TOK = (long)BB * HH * WW_;          // 200704
static constexpr long M_DS  = (long)BB * (HH/2) * (WW_/2);  // 50176

// workspace layout (bytes)
static constexpr size_t OFF_X   = 0;                                   // f32 [M_TOK,128]
static constexpr size_t OFF_XLN = OFF_X   + (size_t)M_TOK*128*4;       // bf16 [M_TOK,128] (reused: [M_DS,512])
static constexpr size_t OFF_QH  = OFF_XLN + (size_t)M_TOK*128*2;       // bf16 [M_TOK,512] (qkv 384 or mlp-h 512)
static constexpr size_t OFF_O   = OFF_QH  + (size_t)M_TOK*512*2;       // bf16 [M_TOK,128]
static constexpr size_t OFF_W   = OFF_O   + (size_t)M_TOK*128*2;       // bf16 weights (524288 elems)

// bf16 TRANSPOSED weight buffer element offsets ([N][K] layout)
static constexpr long WB_QKV  = 0;        // [2][384][128]
static constexpr long WB_PROJ = 98304;    // [2][128][128]
static constexpr long WB_W1   = 131072;   // [2][512][128]
static constexpr long WB_W2   = 262144;   // [2][128][512]
static constexpr long WB_DS   = 393216;   // [256][512]

// ---------------- f32 [K][N] -> bf16 [N][K] transpose-convert (one-time) ----------------
__global__ void cvt_t_kernel(const float* __restrict__ s, bf16_t* __restrict__ d,
                             int K, int N) {
    long i = (long)blockIdx.x * blockDim.x + threadIdx.x;
    if (i < (long)K * N) {
        int k = (int)(i / N), n = (int)(i % N);
        d[(long)n * K + k] = (bf16_t)s[i];
    }
}

// ---------------- fused LayerNorm -> bf16 ----------------
template<int K>
__global__ __launch_bounds__(256) void ln_kernel(const float* __restrict__ x,
                                                 const float* __restrict__ g,
                                                 const float* __restrict__ b,
                                                 bf16_t* __restrict__ y, long M) {
    int lane = threadIdx.x & 31, wv = threadIdx.x >> 5;
    long row = (long)blockIdx.x * 8 + wv;
    if (row >= M) return;
    const float* xr = x + row * K;
    float v[K / 32];
    float s = 0.f;
#pragma unroll
    for (int j = 0; j < K / 32; ++j) { v[j] = xr[lane + j * 32]; s += v[j]; }
    for (int m = 1; m < 32; m <<= 1) s += __shfl_xor(s, m, 32);
    float mean = s * (1.0f / K);
    float q = 0.f;
#pragma unroll
    for (int j = 0; j < K / 32; ++j) { float d = v[j] - mean; q += d * d; }
    for (int m = 1; m < 32; m <<= 1) q += __shfl_xor(q, m, 32);
    float inv = rsqrtf(q * (1.0f / K) + 1e-5f);
#pragma unroll
    for (int j = 0; j < K / 32; ++j) {
        int cc = lane + j * 32;
        y[row * K + cc] = (bf16_t)((v[j] - mean) * inv * g[cc] + b[cc]);
    }
}

// ---------------- 2x2 patch merge + LayerNorm -> bf16 [M_DS,512] ----------------
__global__ __launch_bounds__(256) void merge_ln_kernel(const float* __restrict__ x,
                                                       const float* __restrict__ g,
                                                       const float* __restrict__ b,
                                                       bf16_t* __restrict__ y) {
    int lane = threadIdx.x & 31, wv = threadIdx.x >> 5;
    long row = (long)blockIdx.x * 8 + wv;
    long bi = row / (28 * 28);
    int hw = (int)(row % (28 * 28));
    int h2 = hw / 28, w2 = hw % 28;
    float v[16];
    float s = 0.f;
#pragma unroll
    for (int j = 0; j < 16; ++j) {
        int e = lane + j * 32;                  // 0..511 : (dh,dw,d)
        int dh = e >> 8, dw = (e >> 7) & 1, d = e & 127;
        long src = (((bi * HH) + (2 * h2 + dh)) * WW_ + (2 * w2 + dw)) * (long)DD + d;
        v[j] = x[src];
        s += v[j];
    }
    for (int m = 1; m < 32; m <<= 1) s += __shfl_xor(s, m, 32);
    float mean = s * (1.0f / 512.0f);
    float q = 0.f;
#pragma unroll
    for (int j = 0; j < 16; ++j) { float d = v[j] - mean; q += d * d; }
    for (int m = 1; m < 32; m <<= 1) q += __shfl_xor(q, m, 32);
    float inv = rsqrtf(q * (1.0f / 512.0f) + 1e-5f);
#pragma unroll
    for (int j = 0; j < 16; ++j) {
        int e = lane + j * 32;
        y[row * 512 + e] = (bf16_t)((v[j] - mean) * inv * g[e] + b[e]);
    }
}

// ---------------- WMMA GEMM: C[M,N] = A[M,K] * Bt[N,K]^T (+bias, epilogue) ------
static constexpr int EPI_BF16  = 0;  // +bias, store bf16
static constexpr int EPI_RESID = 1;  // +bias, +resid f32, store f32
static constexpr int EPI_GELU  = 2;  // +bias, gelu, store bf16
static constexpr int EPI_F32   = 3;  // store f32 (no bias)

template<int KTOT, int EPI>
__global__ __launch_bounds__(256) void gemm_kernel(const bf16_t* __restrict__ A,
                                                   const bf16_t* __restrict__ Bt,
                                                   const float* __restrict__ bias,
                                                   const float* __restrict__ resid,
                                                   float* __restrict__ outF,
                                                   bf16_t* __restrict__ outH,
                                                   long M, int N) {
    __shared__ __align__(16) bf16_t As[64][128];
    __shared__ __align__(16) bf16_t Bs[128][136];   // [n][k], +8 pad (row = 272B)

    int t = threadIdx.x;
    int lane = t & 31, wv = t >> 5;
    int lrow = lane & 15, hl = lane >> 4;
    long m0 = (long)blockIdx.x * 64;
    int n0 = blockIdx.y * 128;

    v8f acc[4];
#pragma unroll
    for (int mi = 0; mi < 4; ++mi) acc[mi] = (v8f){0.f,0.f,0.f,0.f,0.f,0.f,0.f,0.f};

    // staging thread mapping
    int ar = t >> 2;              // A: 64 rows, 4 threads/row
    int ac = (t & 3) * 32;
    int br = t >> 1;              // B: 128 rows, 2 threads/row
    int bc = (t & 1) * 64;

    for (int kc = 0; kc < KTOT; kc += 128) {
        const bf16_t* srca = A  + (m0 + ar) * KTOT + kc + ac;
        const bf16_t* srcb = Bt + (long)(n0 + br) * KTOT + kc + bc;
#if HAVE_ASYNC
#pragma unroll
        for (int j = 0; j < 32; j += 8) G2L_B128(srca + j, &As[ar][ac + j]);
#pragma unroll
        for (int j = 0; j < 64; j += 8) G2L_B128(srcb + j, &Bs[br][bc + j]);
        async_wait0();
#else
#pragma unroll
        for (int j = 0; j < 32; j += 8)
            *(uint4*)&As[ar][ac + j] = *(const uint4*)(srca + j);
#pragma unroll
        for (int j = 0; j < 64; j += 8)
            *(uint4*)&Bs[br][bc + j] = *(const uint4*)(srcb + j);
#endif
        __syncthreads();
#pragma unroll
        for (int ks = 0; ks < 4; ++ks) {
            v16bf bfB;
            { // B frag: lane = column n in tile; elems 0..7 = K kh*8.., elems 8..15 = +16
                const bf16_t* pb = &Bs[wv * 16 + lrow][ks * 32 + hl * 8];
                ((uint4*)&bfB)[0] = *(const uint4*)pb;
                ((uint4*)&bfB)[1] = *(const uint4*)(pb + 16);
            }
#pragma unroll
            for (int mi = 0; mi < 4; ++mi) {
                v16bf afA;
                { // A frag: lane = row m; elems 0..7 = K kh*8.., elems 8..15 = +16
                    const bf16_t* pa = &As[mi * 16 + lrow][ks * 32 + hl * 8];
                    ((uint4*)&afA)[0] = *(const uint4*)pa;
                    ((uint4*)&afA)[1] = *(const uint4*)(pa + 16);
                }
                acc[mi] = __builtin_amdgcn_wmma_f32_16x16x32_bf16(
                    false, afA, false, bfB, (short)0, acc[mi], false, false);
            }
        }
        __syncthreads();
    }

    int ncol = n0 + wv * 16 + lrow;
    float bv = 0.f;
    if (EPI != EPI_F32) bv = bias[ncol];
#pragma unroll
    for (int mi = 0; mi < 4; ++mi) {
#pragma unroll
        for (int e = 0; e < 8; ++e) {
            long rowg = m0 + mi * 16 + e + hl * 8;
            long off = rowg * N + ncol;
            float v = acc[mi][e] + bv;
            if (EPI == EPI_BF16) {
                outH[off] = (bf16_t)v;
            } else if (EPI == EPI_RESID) {
                outF[off] = v + resid[off];
            } else if (EPI == EPI_GELU) {
                float gl = 0.5f * v * (1.0f + tanhf(0.7978845608f * (v + 0.044715f * v * v * v)));
                outH[off] = (bf16_t)gl;
            } else {
                outF[off] = v;
            }
        }
    }
}

// ---------------- windowed attention (one workgroup per (batch, window)) -------
template<bool SHIFT>
__global__ __launch_bounds__(256) void attn_kernel(const bf16_t* __restrict__ qkv,
                                                   bf16_t* __restrict__ obuf) {
    __shared__ __align__(16) bf16_t qs[64][DD];        // [tok][d]
    __shared__ __align__(16) bf16_t ksm[64][DD];       // [tok][d]
    __shared__ __align__(16) bf16_t vsm[DD][72];       // [d][tok] (transposed)
    __shared__ __align__(16) bf16_t probs[NHD][64][72];
    __shared__ int rid[64];

    int bi = blockIdx.x / NG;
    int g  = blockIdx.x % NG;
    int g1 = g / NG2, g2 = g % NG2;
    int t = threadIdx.x;

    if (t < 64) {
        int id = 100 + t;                         // pad tokens never match
        if (t < NTOK) {
            int r = t / WS7, c = t % WS7;
            int h = g1 * WS7 + r, w = g2 * WS7 + c;   // post-roll window coords
            int a = (h < HH - WS7) ? 0 : ((h < HH - SH3) ? 1 : 2);
            int bb = (w < WW_ - WS7) ? 0 : ((w < WW_ - SH3) ? 1 : 2);
            id = a * 3 + bb;
        }
        rid[t] = id;
    }

    // gather q,k,v into LDS (v transposed); pad rows 49..63 zero
    {
        int row = t >> 2;
        int col0 = (t & 3) * 32;
        if (row < NTOK) {
            int r = row / WS7, c = row % WS7;
            int h = g1 * WS7 + r, w = g2 * WS7 + c;
            if (SHIFT) { h = (h + SH3) % HH; w = (w + SH3) % WW_; }
            long base = (((long)bi * HH + h) * WW_ + w) * (3 * DD);
#if HAVE_ASYNC
#pragma unroll
            for (int j = 0; j < 32; j += 8) {
                G2L_B128(qkv + base + 0 * DD + col0 + j, &qs[row][col0 + j]);
                G2L_B128(qkv + base + 1 * DD + col0 + j, &ksm[row][col0 + j]);
            }
#endif
#pragma unroll
            for (int j = 0; j < 32; j += 8) {
#if !HAVE_ASYNC
                *(uint4*)&qs[row][col0 + j]  = *(const uint4*)(qkv + base + 0 * DD + col0 + j);
                *(uint4*)&ksm[row][col0 + j] = *(const uint4*)(qkv + base + 1 * DD + col0 + j);
#endif
                union { uint4 q; bf16_t e[8]; } dv;
                dv.q = *(const uint4*)(qkv + base + 2 * DD + col0 + j);
#pragma unroll
                for (int u = 0; u < 8; ++u) vsm[col0 + j + u][row] = dv.e[u];
            }
        } else {
            uint4 z = make_uint4(0u, 0u, 0u, 0u);
#pragma unroll
            for (int j = 0; j < 32; j += 8) {
                *(uint4*)&qs[row][col0 + j]  = z;
                *(uint4*)&ksm[row][col0 + j] = z;
#pragma unroll
                for (int u = 0; u < 8; ++u) vsm[col0 + j + u][row] = (bf16_t)0.f;
            }
        }
    }
    async_wait0();
    __syncthreads();

    int lane = t & 31, wv = t >> 5;
    int hh = wv >> 1;        // head
    int mg = wv & 1;         // which pair of M tiles
    int lrow = lane & 15, hl = lane >> 4;
    const float sc = 0.17677669529663687f;   // 1/sqrt(32)

    // scores = q k^T (per wave: 2 M-tiles x 4 N-tiles, K=32 single step)
    v8f accs[2][4];
#pragma unroll
    for (int mi2 = 0; mi2 < 2; ++mi2) {
        int mi = mg * 2 + mi2;
        v16bf afQ;
        {
            const bf16_t* pa = &qs[mi * 16 + lrow][hh * HD + hl * 8];
            ((uint4*)&afQ)[0] = *(const uint4*)pa;
            ((uint4*)&afQ)[1] = *(const uint4*)(pa + 16);
        }
#pragma unroll
        for (int ni = 0; ni < 4; ++ni) {
            v16bf bfK;
            const bf16_t* pb = &ksm[ni * 16 + lrow][hh * HD + hl * 8];
            ((uint4*)&bfK)[0] = *(const uint4*)pb;
            ((uint4*)&bfK)[1] = *(const uint4*)(pb + 16);
            v8f cz = (v8f){0.f,0.f,0.f,0.f,0.f,0.f,0.f,0.f};
            accs[mi2][ni] = __builtin_amdgcn_wmma_f32_16x16x32_bf16(
                false, afQ, false, bfK, (short)0, cz, false, false);
        }
    }

    // scale + mask + row softmax (row lives in 16-lane half across 4 N tiles)
#pragma unroll
    for (int mi2 = 0; mi2 < 2; ++mi2) {
        int mi = mg * 2 + mi2;
#pragma unroll
        for (int e = 0; e < 8; ++e) {
            int rtok = mi * 16 + e + hl * 8;
            int myid = rid[rtok];
            float vals[4];
            float rmax = -1e30f;
#pragma unroll
            for (int ni = 0; ni < 4; ++ni) {
                int ctok = ni * 16 + lrow;
                float v = accs[mi2][ni][e] * sc;
                if (ctok >= NTOK) v = -1e30f;
                else if (SHIFT && rid[ctok] != myid) v += -100.0f;
                vals[ni] = v;
                rmax = fmaxf(rmax, v);
            }
            for (int m = 1; m <= 8; m <<= 1) rmax = fmaxf(rmax, __shfl_xor(rmax, m, 32));
            float s = 0.f;
#pragma unroll
            for (int ni = 0; ni < 4; ++ni) { vals[ni] = __expf(vals[ni] - rmax); s += vals[ni]; }
            for (int m = 1; m <= 8; m <<= 1) s += __shfl_xor(s, m, 32);
            float inv = 1.0f / s;
#pragma unroll
            for (int ni = 0; ni < 4; ++ni)
                probs[hh][rtok][ni * 16 + lrow] = (bf16_t)(vals[ni] * inv);
        }
    }
    __syncthreads();

    // o = p @ v  (per wave: 2 M-tiles x 2 d-tiles, K = 64 in 2 steps)
#pragma unroll
    for (int mi2 = 0; mi2 < 2; ++mi2) {
        int mi = mg * 2 + mi2;
#pragma unroll
        for (int ni = 0; ni < 2; ++ni) {
            v8f acc = (v8f){0.f,0.f,0.f,0.f,0.f,0.f,0.f,0.f};
#pragma unroll
            for (int ksn = 0; ksn < 2; ++ksn) {
                v16bf afP;
                const bf16_t* pa = &probs[hh][mi * 16 + lrow][ksn * 32 + hl * 8];
                ((uint4*)&afP)[0] = *(const uint4*)pa;
                ((uint4*)&afP)[1] = *(const uint4*)(pa + 16);
                v16bf bfV;
                const bf16_t* pb = &vsm[hh * HD + ni * 16 + lrow][ksn * 32 + hl * 8];
                ((uint4*)&bfV)[0] = *(const uint4*)pb;
                ((uint4*)&bfV)[1] = *(const uint4*)(pb + 16);
                acc = __builtin_amdgcn_wmma_f32_16x16x32_bf16(
                    false, afP, false, bfV, (short)0, acc, false, false);
            }
#pragma unroll
            for (int e = 0; e < 8; ++e) {
                int rtok = mi * 16 + e + hl * 8;
                if (rtok < NTOK) {
                    int r = rtok / WS7, c = rtok % WS7;
                    int h = g1 * WS7 + r, w = g2 * WS7 + c;
                    if (SHIFT) { h = (h + SH3) % HH; w = (w + SH3) % WW_; }
                    long o = (((long)bi * HH + h) * WW_ + w) * DD + hh * HD + ni * 16 + lrow;
                    obuf[o] = (bf16_t)acc[e];
                }
            }
        }
    }
}

// ---------------- host side ----------------
extern "C" void kernel_launch(void* const* d_in, const int* in_sizes, int n_in,
                              void* d_out, int out_size, void* d_ws, size_t ws_size,
                              hipStream_t stream) {
    const float* x      = (const float*)d_in[0];
    const float* ln1_g  = (const float*)d_in[1];
    const float* ln1_b  = (const float*)d_in[2];
    const float* qkv_w  = (const float*)d_in[3];
    const float* qkv_b  = (const float*)d_in[4];
    const float* proj_w = (const float*)d_in[5];
    const float* proj_b = (const float*)d_in[6];
    const float* ln2_g  = (const float*)d_in[7];
    const float* ln2_b  = (const float*)d_in[8];
    const float* mlp_w1 = (const float*)d_in[9];
    const float* mlp_b1 = (const float*)d_in[10];
    const float* mlp_w2 = (const float*)d_in[11];
    const float* mlp_b2 = (const float*)d_in[12];
    const float* ds_g   = (const float*)d_in[13];
    const float* ds_b   = (const float*)d_in[14];
    const float* ds_w   = (const float*)d_in[15];

    char* ws = (char*)d_ws;
    float*  x_cur = (float*)(ws + OFF_X);
    bf16_t* xln   = (bf16_t*)(ws + OFF_XLN);
    bf16_t* qh    = (bf16_t*)(ws + OFF_QH);
    bf16_t* obuf  = (bf16_t*)(ws + OFF_O);
    bf16_t* wb    = (bf16_t*)(ws + OFF_W);

    (void)hipMemcpyAsync(x_cur, x, (size_t)M_TOK * 128 * 4, hipMemcpyDeviceToDevice, stream);

    // one-time transpose-convert of all weights to bf16 [N][K]
    for (int i = 0; i < 2; ++i) {
        cvt_t_kernel<<<(49152 + 255) / 256, 256, 0, stream>>>(
            qkv_w + (long)i * 49152,  wb + WB_QKV  + (long)i * 49152, 128, 384);
        cvt_t_kernel<<<(16384 + 255) / 256, 256, 0, stream>>>(
            proj_w + (long)i * 16384, wb + WB_PROJ + (long)i * 16384, 128, 128);
        cvt_t_kernel<<<(65536 + 255) / 256, 256, 0, stream>>>(
            mlp_w1 + (long)i * 65536, wb + WB_W1   + (long)i * 65536, 128, 512);
        cvt_t_kernel<<<(65536 + 255) / 256, 256, 0, stream>>>(
            mlp_w2 + (long)i * 65536, wb + WB_W2   + (long)i * 65536, 512, 128);
    }
    cvt_t_kernel<<<(131072 + 255) / 256, 256, 0, stream>>>(ds_w, wb + WB_DS, 512, 256);

    const int MBLK = (int)(M_TOK / 64);   // 3136
    const int LNB  = (int)(M_TOK / 8);    // 25088

    for (int i = 0; i < 2; ++i) {
        // LN1 -> bf16
        ln_kernel<128><<<LNB, 256, 0, stream>>>(x_cur, ln1_g + i * 128, ln1_b + i * 128, xln, M_TOK);
        // QKV: [M,128] x [128,384] + b -> bf16
        gemm_kernel<128, EPI_BF16><<<dim3(MBLK, 3), 256, 0, stream>>>(
            xln, wb + WB_QKV + (long)i * 49152, qkv_b + i * 384, nullptr, nullptr, qh, M_TOK, 384);
        // windowed attention (shift folded into indices for layer 1)
        if (i == 0) attn_kernel<false><<<BB * NG, 256, 0, stream>>>(qh, obuf);
        else        attn_kernel<true ><<<BB * NG, 256, 0, stream>>>(qh, obuf);
        // proj + bias + residual -> f32 x_cur
        gemm_kernel<128, EPI_RESID><<<dim3(MBLK, 1), 256, 0, stream>>>(
            obuf, wb + WB_PROJ + (long)i * 16384, proj_b + i * 128, x_cur, x_cur, nullptr, M_TOK, 128);
        // LN2 -> bf16
        ln_kernel<128><<<LNB, 256, 0, stream>>>(x_cur, ln2_g + i * 128, ln2_b + i * 128, xln, M_TOK);
        // MLP1 + bias + GELU -> bf16 hidden
        gemm_kernel<128, EPI_GELU><<<dim3(MBLK, 4), 256, 0, stream>>>(
            xln, wb + WB_W1 + (long)i * 65536, mlp_b1 + i * 512, nullptr, nullptr, qh, M_TOK, 512);
        // MLP2 + bias + residual -> f32 x_cur
        gemm_kernel<512, EPI_RESID><<<dim3(MBLK, 1), 256, 0, stream>>>(
            qh, wb + WB_W2 + (long)i * 65536, mlp_b2 + i * 128, x_cur, x_cur, nullptr, M_TOK, 128);
    }

    // downsample: 2x2 merge + LN -> bf16 [M_DS,512], then x ds_w -> f32 out
    merge_ln_kernel<<<(int)(M_DS / 8), 256, 0, stream>>>(x_cur, ds_g, ds_b, xln);
    gemm_kernel<512, EPI_F32><<<dim3((int)(M_DS / 64), 2), 256, 0, stream>>>(
        xln, wb + WB_DS, nullptr, nullptr, (float*)d_out, nullptr, M_DS, 256);
}
